// GetAlignedFeature_9234179686716
// MI455X (gfx1250) — matrix operand order repaired and checked by reading, not compiled, not run on hardware
//
#include <hip/hip_runtime.h>
#include <stdint.h>

#define B_   4
#define C_   1024
#define H_   64
#define W_   64
#define HW_  4096
#define OUT_ 2048
#define K_   1024

#define BM 128
#define BN 128
#define BK 32

#define NSLICE 8            // offset-conv channel slices (2*C / 256)

typedef __attribute__((ext_vector_type(16))) __bf16 v16bf;
typedef __attribute__((ext_vector_type(8)))  float  v8f;

__device__ __forceinline__ unsigned short f2bf(float f) {
    unsigned u = __float_as_uint(f);
    unsigned r = (u + 0x7FFFu + ((u >> 16) & 1u)) >> 16;   // round-to-nearest-even
    return (unsigned short)r;
}

// ---------------------------------------------------------------------------
// Kernel A: convert w_def (f32 [2048][1024]) -> bf16 (one-shot, NT loads)
// ---------------------------------------------------------------------------
__global__ void cvt_wdef_bf16(const float* __restrict__ w,
                              unsigned short* __restrict__ o, int n) {
    int i = blockIdx.x * blockDim.x + threadIdx.x;
    if (i < n) o[i] = f2bf(__builtin_nontemporal_load(w + i));
}

// ---------------------------------------------------------------------------
// Kernel B1: 3x3 conv (2048 -> 2 ch, pad=1), channel-sliced partial sums.
// Block = (slice, b, h); lane = w (coalesced along x). 2048 blocks.
// Deterministic: each slice writes its own partial buffer (no atomics).
// ---------------------------------------------------------------------------
__global__ void offset_conv_slice(const float* __restrict__ refer,
                                  const float* __restrict__ sup,
                                  const float* __restrict__ woff,   // [2][2048][3][3]
                                  float* __restrict__ partial)      // [NSLICE][B][2][HW]
{
    int blk   = blockIdx.x;            // 0..2047
    int slice = blk >> 8;              // 0..7
    int bh    = blk & 255;
    int b     = bh >> 6;
    int h     = bh & 63;
    int w     = threadIdx.x;           // 0..63
    float acc0 = 0.0f, acc1 = 0.0f;
    int ci0 = slice * (2 * C_ / NSLICE);      // 256 channels per slice
    for (int ci = ci0; ci < ci0 + (2 * C_ / NSLICE); ++ci) {
        const float* fp = (ci < C_)
            ? (refer + (size_t)(b * C_ + ci) * HW_)
            : (sup   + (size_t)(b * C_ + (ci - C_)) * HW_);
        const float* w0 = woff + (size_t)ci * 9;
        const float* w1 = woff + (size_t)(2 * C_ + ci) * 9;
        #pragma unroll
        for (int ky = 0; ky < 3; ++ky) {
            int yy = h + ky - 1;
            bool yok = (yy >= 0) && (yy < H_);
            const float* row = fp + yy * W_;
            #pragma unroll
            for (int kx = 0; kx < 3; ++kx) {
                int xx = w + kx - 1;
                float v = (yok && xx >= 0 && xx < W_) ? row[xx] : 0.0f;
                acc0 = fmaf(v, w0[ky * 3 + kx], acc0);
                acc1 = fmaf(v, w1[ky * 3 + kx], acc1);
            }
        }
    }
    int l = h * W_ + w;
    partial[(((size_t)slice * B_ + b) * 2 + 0) * HW_ + l] = acc0;
    partial[(((size_t)slice * B_ + b) * 2 + 1) * HW_ + l] = acc1;
}

// ---------------------------------------------------------------------------
// Kernel B2: reduce slice partials + bias -> off[B][2][HW]
// ---------------------------------------------------------------------------
__global__ void offset_reduce(const float* __restrict__ partial,
                              const float* __restrict__ boff,
                              float* __restrict__ off)
{
    int i  = blockIdx.x * blockDim.x + threadIdx.x;   // 0 .. B*2*HW-1
    int ch = (i >> 12) & 1;                           // [b][2][HW] layout
    float s = boff[ch];
    #pragma unroll
    for (int sl = 0; sl < NSLICE; ++sl)
        s += partial[(size_t)sl * (B_ * 2 * HW_) + i];
    off[i] = s;
}

// ---------------------------------------------------------------------------
// Kernel C: bilinear deform-sample sup_feat at (h+dy, w+dx), zero padded.
// Lanes = pixels; stores sampled[b][c][l] lane-contiguous (coalesced bf16).
// ---------------------------------------------------------------------------
__global__ void deform_sample(const float* __restrict__ sup,
                              const float* __restrict__ off,
                              unsigned short* __restrict__ samp) // [B][C][HW] bf16
{
    int cs = blockIdx.x >> 6;                 // channel slice 0..7 (128 ch each)
    int pb = blockIdx.x & 63;                 // pixel block
    int p  = pb * 256 + threadIdx.x;          // 0..16383
    int b  = p >> 12;
    int l  = p & (HW_ - 1);
    int h  = l >> 6;
    int w  = l & 63;

    float dy = off[((size_t)b * 2 + 0) * HW_ + l];
    float dx = off[((size_t)b * 2 + 1) * HW_ + l];
    float py = (float)h + dy;
    float px = (float)w + dx;
    float y0 = floorf(py), x0 = floorf(px);
    float wy = py - y0,    wx = px - x0;
    int y0i = (int)y0, x0i = (int)x0;
    int y1i = y0i + 1,  x1i = x0i + 1;

    int   i00, i01, i10, i11;
    float w00, w01, w10, w11;
    auto mk = [&](int yi, int xi, float wgt, int& idx, float& wv) {
        bool ok = (yi >= 0) && (yi < H_) && (xi >= 0) && (xi < W_);
        int yc = min(max(yi, 0), H_ - 1);
        int xc = min(max(xi, 0), W_ - 1);
        idx = yc * W_ + xc;
        wv  = ok ? wgt : 0.0f;
    };
    mk(y0i, x0i, (1.0f - wy) * (1.0f - wx), i00, w00);
    mk(y0i, x1i, (1.0f - wy) * wx,          i01, w01);
    mk(y1i, x0i, wy * (1.0f - wx),          i10, w10);
    mk(y1i, x1i, wy * wx,                   i11, w11);

    const float* base = sup + (size_t)b * C_ * HW_;
    unsigned short* sb = samp + (size_t)b * C_ * HW_ + l;
    int c0 = cs * 128;
    for (int c = c0; c < c0 + 128; ++c) {
        const float* fp = base + (size_t)c * HW_;
        float s = w00 * fp[i00] + w01 * fp[i01] + w10 * fp[i10] + w11 * fp[i11];
        sb[(size_t)c * HW_] = f2bf(s);
    }
}

// ---------------------------------------------------------------------------
// Kernel D: batched GEMM  out[b] = w_def(2048x1024) x sampled[b](1024x4096)
// bf16 operands, f32 accumulate via v_wmma_f32_16x16x32_bf16.
// 256 threads = 8 waves (2x4), block tile 128x128, K-step 32.
// Double-buffered LDS -> ONE barrier per K-iteration; next-tile global
// loads issued before the WMMA block (VMEM latency hidden behind matrix
// math); s_wait_loadcnt only lands after the WMMAs, ahead of the LDS drain.
// Chunk-XOR swizzle keeps b128 fragment loads bank-conflict free.
// Output stored nontemporal so the 128MB result doesn't evict the
// L2-resident bf16 operands (192MB L2 vs 36MB of operands).
// ---------------------------------------------------------------------------
__global__ void __launch_bounds__(256)
deform_gemm(const unsigned short* __restrict__ wdef,  // [2048][1024] bf16
            const unsigned short* __restrict__ samp,  // [B][1024][4096] bf16
            float* __restrict__ out)                  // [B][2048][4096] f32
{
    __shared__ unsigned short As[2][BM * BK];  // [m][k], 16B chunks XOR-swizzled
    __shared__ unsigned short Bs[2][BN * BK];  // [n][k] (transposed), swizzled

    const int tid  = threadIdx.x;
    const int lane = tid & 31;
    const int wid  = tid >> 5;
    const int l16  = lane & 15;
    const int lh   = lane >> 4;

    const int mbN = OUT_ / BM;   // 16
    const int nbN = HW_  / BN;   // 32
    int blk = blockIdx.x;
    const int b   = blk / (mbN * nbN);
    int rem       = blk % (mbN * nbN);
    const int blockM = (rem / nbN) * BM;
    const int blockN = (rem % nbN) * BN;

    const unsigned short* aG = wdef + (size_t)blockM * K_;
    const unsigned short* bG = samp + (size_t)b * C_ * HW_ + blockN;

    // staging indices
    const int ar  = tid >> 1;          // A row 0..127
    const int ac0 = (tid & 1) * 2;     // A 16B-chunk pair {ac0, ac0+1} of 4
    const int bkr = tid >> 3;          // B k-row 0..31
    const int bs0 = (tid & 7) * 2;     // B col-seg pair of 16 (8 cols each)

    const int waveM = (wid >> 2) * 64; // 0,64
    const int waveN = (wid & 3) * 32;  // 0,32,64,96

    uint4 pa[2], pb[2];
    auto loadG = [&](int kb) {
        #pragma unroll
        for (int cc = 0; cc < 2; ++cc)
            pa[cc] = *(const uint4*)(aG + (size_t)ar * K_ + kb + (ac0 + cc) * 8);
        #pragma unroll
        for (int ss = 0; ss < 2; ++ss)
            pb[ss] = *(const uint4*)(bG + (size_t)(kb + bkr) * HW_ + (bs0 + ss) * 8);
    };
    auto storeL = [&](int buf) {
        #pragma unroll
        for (int cc = 0; cc < 2; ++cc) {
            int ch  = ac0 + cc;
            int chs = ch ^ (ar & 3);
            *(uint4*)&As[buf][ar * BK + chs * 8] = pa[cc];
        }
        #pragma unroll
        for (int ss = 0; ss < 2; ++ss) {
            int seg = bs0 + ss;
            const unsigned short* e = (const unsigned short*)&pb[ss];
            #pragma unroll
            for (int j = 0; j < 8; ++j) {
                int n   = seg * 8 + j;
                int chs = ((bkr >> 3) ^ (n & 3));
                Bs[buf][n * BK + chs * 8 + (bkr & 7)] = e[j];
            }
        }
    };

    v8f acc[4][2] = {};

    loadG(0);
    storeL(0);
    int cur = 0;

    for (int kb = 0; kb < K_; kb += BK) {
        __syncthreads();   // publish buf[cur]; retire last iteration's stores
        bool more = (kb + BK) < K_;
        if (more) loadG(kb + BK);        // in flight during the WMMA block

        // ---- fragments from buf[cur] (ISA 7.12.2 layouts) ----
        union FU { uint4 q[2]; v16bf v; };
        FU a[4], bfr[2];
        #pragma unroll
        for (int mt = 0; mt < 4; ++mt) {
            int m  = waveM + mt * 16 + l16;
            int c0 = (lh)     ^ (m & 3);        // v0..3: k = lh*8 + 0..7
            int c1 = (2 + lh) ^ (m & 3);        // v4..7: k = 16 + lh*8 + 0..7
            a[mt].q[0] = *(const uint4*)&As[cur][m * BK + c0 * 8];
            a[mt].q[1] = *(const uint4*)&As[cur][m * BK + c1 * 8];
        }
        #pragma unroll
        for (int nt = 0; nt < 2; ++nt) {
            int n  = waveN + nt * 16 + l16;
            int c0 = (lh)     ^ (n & 3);        // v0..3: k = lh*8 + 0..7
            int c1 = (2 + lh) ^ (n & 3);        // v4..7: k = 16 + lh*8 + 0..7
            bfr[nt].q[0] = *(const uint4*)&Bs[cur][n * BK + c0 * 8];
            bfr[nt].q[1] = *(const uint4*)&Bs[cur][n * BK + c1 * 8];
        }
        #pragma unroll
        for (int mt = 0; mt < 4; ++mt)
            #pragma unroll
            for (int nt = 0; nt < 2; ++nt)
                acc[mt][nt] = __builtin_amdgcn_wmma_f32_16x16x32_bf16(
                    false, a[mt].v, false, bfr[nt].v,
                    (short)0, acc[mt][nt], false, false);

        if (more) storeL(cur ^ 1);       // writer side of the double buffer
        cur ^= 1;
    }

    // ---- store D nontemporal (VGPR v: M = v + lh*8, N = l16) ----
    float* oB = out + ((size_t)b * OUT_ + blockM) * HW_ + blockN;
    #pragma unroll
    for (int mt = 0; mt < 4; ++mt)
        #pragma unroll
        for (int nt = 0; nt < 2; ++nt)
            #pragma unroll
            for (int v = 0; v < 8; ++v) {
                int o = waveM + mt * 16 + lh * 8 + v;
                int n = waveN + nt * 16 + l16;
                __builtin_nontemporal_store(acc[mt][nt][v],
                                            oB + (size_t)o * HW_ + n);
            }
}

// ---------------------------------------------------------------------------
extern "C" void kernel_launch(void* const* d_in, const int* in_sizes, int n_in,
                              void* d_out, int out_size, void* d_ws, size_t ws_size,
                              hipStream_t stream) {
    const float* refer = (const float*)d_in[0];
    const float* sup   = (const float*)d_in[1];
    const float* woff  = (const float*)d_in[2];
    const float* boff  = (const float*)d_in[3];
    const float* wdef  = (const float*)d_in[4];
    float* out = (float*)d_out;

    // workspace layout (all 16B aligned)
    char* ws = (char*)d_ws;
    float*          off     = (float*)ws;                               // 128 KB
    unsigned short* wdef_bf = (unsigned short*)(ws + 131072);           // 4 MB
    unsigned short* samp    = (unsigned short*)(ws + 131072 + 4194304); // 32 MB
    float*          part    = (float*)(ws + 131072 + 4194304 + 33554432); // 1 MB

    cvt_wdef_bf16    <<<(OUT_ * K_) / 256, 256, 0, stream>>>(wdef, wdef_bf, OUT_ * K_);
    offset_conv_slice<<<NSLICE * B_ * H_, 64, 0, stream>>>(refer, sup, woff, part);
    offset_reduce    <<<(B_ * 2 * HW_) / 256, 256, 0, stream>>>(part, boff, off);
    deform_sample    <<<512, 256, 0, stream>>>(sup, off, samp);
    deform_gemm      <<<B_ * (OUT_ / BM) * (HW_ / BN), 256, 0, stream>>>(wdef_bf, samp, out);
}
